// StoryJudger2_86629490360613
// MI455X (gfx1250) — matrix operand order repaired and checked by reading, not compiled, not run on hardware
//
#include <hip/hip_runtime.h>
#include <math.h>

// ---------------------------------------------------------------------------
// StoryJudger2 on MI455X (gfx1250, wave32, WMMA + TDM).
//   B=8, S=256, V=50257, E=512, P=768, H=1000.
//   Padded dims: Hp=1024, Np=4*Hp=4096 (gate blocks 1000 -> 1024), M=B*S=2048.
// Precision: bf16 operands, f32 accumulation (v_wmma_f32_16x16x32_bf16).
// ---------------------------------------------------------------------------

typedef __attribute__((ext_vector_type(16))) __bf16 v16bf;
typedef __attribute__((ext_vector_type(8)))  float  v8f;
typedef __attribute__((ext_vector_type(4)))  unsigned int u32x4;
typedef __attribute__((ext_vector_type(8)))  int    i32x8;
typedef __attribute__((ext_vector_type(4)))  int    i32x4;

#define BB   8
#define SS   256
#define MM   2048      // B*S
#define HP   1024
#define NP   4096      // 4*HP
#define NT   256       // NP/16  (N tiles)
#define MT   128       // MM/16  (M tiles)

__device__ __forceinline__ float bf2f(unsigned short u) {
  unsigned int x = ((unsigned int)u) << 16;
  return __builtin_bit_cast(float, x);
}
__device__ __forceinline__ unsigned short f2bf(float f) {
  unsigned int x = __builtin_bit_cast(unsigned int, f);
  unsigned int r = x + 0x7FFFu + ((x >> 16) & 1u);   // RNE
  return (unsigned short)(r >> 16);
}
__device__ __forceinline__ float sigm(float x) {
  return 1.0f / (1.0f + __expf(-x));
}

// ---------------------------------------------------------------------------
// 1-D Tensor Data Mover copy: `bytes` (multiple of 8) from global -> LDS.
// Builds a D# (CDNA5 ISA): group0 = {count=1, lds_addr,
// global_addr, type=2}; group1 = {data_size=8B, tensor_dim0=n8, tile_dim0=n8,
// tile_dim1=1, stride0=n8}.  Tracked by TENSORcnt.
// ---------------------------------------------------------------------------
__device__ __forceinline__ void tdm_load_1d(unsigned ldsOff, const void* gsrc,
                                            unsigned bytes) {
  unsigned long long ga = (unsigned long long)gsrc;
  unsigned n8 = bytes >> 3;
  u32x4 g0;
  g0[0] = 1u;                                            // count=1
  g0[1] = ldsOff;                                        // lds_addr
  g0[2] = (unsigned)(ga & 0xFFFFFFFFu);                  // global_addr lo
  g0[3] = (unsigned)((ga >> 32) & 0x01FFFFFFu) | (2u << 30);  // hi | type=2
  i32x8 g1;
  g1[0] = (3 << 16);                                     // data_size = 8B
  g1[1] = (int)((n8 & 0xFFFFu) << 16);                   // tensor_dim0 lo16
  g1[2] = (int)((n8 >> 16) & 0xFFFFu) | (1 << 16);       // dim0 hi | dim1=1 lo
  g1[3] = (int)((n8 & 0xFFFFu) << 16);                   // dim1 hi=0 | tile_dim0
  g1[4] = 1;                                             // tile_dim1=1, tile_dim2=0
  g1[5] = (int)n8;                                       // stride0 lo32
  g1[6] = 0;                                             // stride0 hi | stride1 lo
  g1[7] = 0;
  i32x4 z = {};
#if __clang_major__ >= 23
  i32x8 z8 = {};
  __builtin_amdgcn_tensor_load_to_lds(g0, g1, z, z, z8, 0);
#else
  __builtin_amdgcn_tensor_load_to_lds(g0, g1, z, z, 0);
#endif
}

// ---------------------------------------------------------------------------
// Embedding gather:  out[row, col] = bf16(table[x[row]*K + col]),  row-major.
// ---------------------------------------------------------------------------
__global__ void k_gather(const int* __restrict__ x,
                         const float* __restrict__ table,
                         unsigned short* __restrict__ out, int K) {
  int col = blockIdx.x * 256 + threadIdx.x;
  int row = blockIdx.y;
  if (col < K) {
    int idx = x[row];
    out[(size_t)row * K + col] = f2bf(table[(size_t)idx * K + col]);
  }
}

// ---------------------------------------------------------------------------
// Pack activations (bf16 row-major M x Kp) into WMMA A-tile layout.
// A 16x32 bf16 tile (ISA 7.12.2): lane l holds row M=l%16;
//   element e -> K%32 = (e%8) + 16*(e/8) + 8*(l/16).
// Tile order: [mt][kt], 512 bf16 per tile.
// ---------------------------------------------------------------------------
__global__ void k_pack_a(const unsigned short* __restrict__ src,
                         unsigned short* __restrict__ dst, int Kp, int KT,
                         int total) {
  int g = blockIdx.x * 256 + threadIdx.x;
  if (g >= total) return;
  int e    = g & 15;
  int lane = (g >> 4) & 31;
  int tile = g >> 9;
  int kt = tile % KT;
  int mt = tile / KT;
  int M  = mt * 16 + (lane & 15);
  int K  = kt * 32 + (e & 7) + 16 * (e >> 3) + 8 * (lane >> 4);
  dst[g] = src[(size_t)M * Kp + K];
}

// ---------------------------------------------------------------------------
// Pack weights W (f32, row-major Norig x Korig) into WMMA B-tile layout for
// B = W^T (Kp x NP).  B 32x16 bf16 tile: lane l holds col N=l%16,
//   element e -> K%32 = (e%8) + 16*(e/8) + 8*(l/16).
// N remap: 4000 rows -> 4096 (4 gate blocks of 1000 -> 1024, zero pad).
// K remap: blocks of kblkO -> kblkP (identity when equal), zero pad.
// Tile order: [kt][nt], 512 bf16 per tile.
// ---------------------------------------------------------------------------
__global__ void k_pack_b(const float* __restrict__ W,
                         unsigned short* __restrict__ dst,
                         int Korig, int kblkO, int kblkP, int total) {
  int g = blockIdx.x * 256 + threadIdx.x;
  if (g >= total) return;
  int e    = g & 15;
  int lane = (g >> 4) & 31;
  int tile = g >> 9;
  int nt = tile % NT;
  int kt = tile / NT;
  int N  = nt * 16 + (lane & 15);
  int K  = kt * 32 + (e & 7) + 16 * (e >> 3) + 8 * (lane >> 4);
  int nb = N >> 10, no = N & 1023;
  int kb = K / kblkP, ko = K % kblkP;
  unsigned short v = 0;
  if (no < 1000 && ko < kblkO) {
    int Nor = nb * 1000 + no;
    int Kor = kb * kblkO + ko;
    if (Kor < Korig) v = f2bf(W[(size_t)Nor * Korig + Kor]);
  }
  dst[g] = v;
}

// bias (4000) -> padded 4096 f32
__global__ void k_pack_bias(const float* __restrict__ b, float* __restrict__ o) {
  int i = blockIdx.x * 256 + threadIdx.x;
  int nb = i >> 10, no = i & 1023;
  o[i] = (no < 1000) ? b[nb * 1000 + no] : 0.0f;
}

// ---------------------------------------------------------------------------
// Input-projection GEMM:  xg[M, NP] = Apack(M x Kp, bf16) * Bpack + bias.
// Block = 256 threads (8 waves); block tile 64M x 128N; wave tile 16M x 64N.
// B K-slices (8 contiguous tiles = 8 KB) are double-buffered through LDS by
// the Tensor Data Mover (wave 0 issues, s_wait_tensorcnt + barrier handshake),
// overlapping DMA of slice kt+1 with WMMAs on slice kt and giving 4x reuse.
// ---------------------------------------------------------------------------
__global__ void __launch_bounds__(256)
k_gemm(const unsigned short* __restrict__ Apack,
       const unsigned short* __restrict__ Bpack,
       const float* __restrict__ bias,
       float* __restrict__ out, int KT) {
  const int lane = threadIdx.x & 31;
  const int wv   = threadIdx.x >> 5;
  const int mt   = blockIdx.y * 4 + (wv >> 1);
  const int ntb0 = blockIdx.x * 8;          // block's first N tile
  const int jb   = (wv & 1) * 4;            // wave's 4-tile offset in slice

  __shared__ __align__(32) unsigned short bbuf[2][8 * 512];  // 2 x 8 KB

  const unsigned short* bsrc = Bpack + (size_t)ntb0 * 512;   // +kt*NT*512
  if (wv == 0) {
    tdm_load_1d((unsigned)(unsigned long long)&bbuf[0][0], bsrc, 8192);
    __builtin_amdgcn_s_wait_tensorcnt(0);
  }
  __syncthreads();

  const unsigned short* alane = Apack + ((size_t)mt * KT) * 512 + lane * 16;
  v8f acc[4] = {};
  for (int kt = 0; kt < KT; ++kt) {
    if (wv == 0 && kt + 1 < KT)
      tdm_load_1d((unsigned)(unsigned long long)&bbuf[(kt + 1) & 1][0],
                  bsrc + (size_t)(kt + 1) * (NT * 512), 8192);
    const v16bf a = *(const v16bf*)(alane + (size_t)kt * 512);
    const unsigned short* bl = &bbuf[kt & 1][jb * 512 + lane * 16];
#pragma unroll
    for (int j = 0; j < 4; ++j) {
      const v16bf b = *(const v16bf*)(bl + j * 512);
      acc[j] = __builtin_amdgcn_wmma_f32_16x16x32_bf16(
          false, a, false, b, (short)0, acc[j], false, false);
    }
    if (wv == 0 && kt + 1 < KT) __builtin_amdgcn_s_wait_tensorcnt(0);
    __syncthreads();
  }
  // D layout: VGPR r, lane l -> M=r+8*(l/16), N=l%16
  const int row0 = mt * 16 + 8 * (lane >> 4);
#pragma unroll
  for (int j = 0; j < 4; ++j) {
    int col = (ntb0 + jb + j) * 16 + (lane & 15);
    float bv = bias[col];
#pragma unroll
    for (int r = 0; r < 8; ++r)
      out[(size_t)(row0 + r) * NP + col] = acc[j][r] + bv;
  }
}

// ---------------------------------------------------------------------------
// Persistent LSTM recurrence: one 1024-thread workgroup (32 waves) per
// direction (blockIdx.x).  h (16 x HP, rows 8..15 zero pad) lives in LDS in
// WMMA-A layout; Whh streams from L2 as packed B tiles (8 MB, L2-resident;
// all six direction weights = 48 MB < 192 MB L2).  Per step: g = h @ Whh^T
// (8192 wmma/WGP) + xg[t]; fused gate math; h/c update; hs store.
// Wave wv owns hidden cols [32*wv, 32*wv+32) in all 4 gate regions, so the
// i/f/g/o mix needs no cross-wave traffic.  The 8 B tiles per K-step are at
// compile-time byte offsets from one wave pointer (fold into the 24-bit
// instruction offset -> no per-load address math).
// ---------------------------------------------------------------------------
__global__ void __launch_bounds__(1024)
k_lstm(const float* __restrict__ xg0, const float* __restrict__ xg1,
       const unsigned short* __restrict__ Whh0,
       const unsigned short* __restrict__ Whh1,
       unsigned short* __restrict__ hsOut, int outStride) {
  const int dir  = blockIdx.x;
  const float* xg = dir ? xg1 : xg0;
  const unsigned short* Whh = dir ? Whh1 : Whh0;
  const int colOff = dir ? HP : 0;

  const int tid  = threadIdx.x;
  const int lane = tid & 31;
  const int wv   = tid >> 5;
  const int nn   = lane & 15;
  const int hiM  = 8 * (lane >> 4);

  __shared__ __align__(32) unsigned short hpack[32 * 512];  // 32 KB, A-layout
  for (int i = tid; i < 32 * 512; i += 1024) hpack[i] = 0;
  __syncthreads();

  // wave-base pointer: tiles (G, sub) at constant offsets G*64*512 + sub*512
  const unsigned short* wlane = Whh + (size_t)(wv * 2) * 512 + lane * 16;

  v8f c0 = {}, c1 = {};

  for (int s = 0; s < SS; ++s) {
    const int tpos = dir ? (SS - 1 - s) : s;

    v8f acc[8] = {};
    for (int kt = 0; kt < 32; ++kt) {
      const v16bf a = *(const v16bf*)&hpack[kt * 512 + lane * 16];
      const unsigned short* p = wlane + (size_t)kt * (NT * 512);
      if (kt + 1 < 32) __builtin_prefetch(p + NT * 512, 0, 1);
#pragma unroll
      for (int j = 0; j < 8; ++j) {
        const v16bf b = *(const v16bf*)(p + (j >> 1) * (64 * 512) + (j & 1) * 512);
        acc[j] = __builtin_amdgcn_wmma_f32_16x16x32_bf16(
            false, a, false, b, (short)0, acc[j], false, false);
      }
    }
    __syncthreads();  // all waves finished reading h before it is rewritten

#pragma unroll
    for (int s2 = 0; s2 < 2; ++s2) {
#pragma unroll
      for (int r = 0; r < 8; ++r) {
        const int M    = r + hiM;              // padded batch row
        const int hcol = wv * 32 + s2 * 16 + nn;
        float xi = 0.f, xf = 0.f, xc = 0.f, xo = 0.f;
        if (M < BB) {
          const float* xr = xg + ((size_t)(M * SS + tpos)) * NP;
          xi = xr[0 * HP + hcol];
          xf = xr[1 * HP + hcol];
          xc = xr[2 * HP + hcol];
          xo = xr[3 * HP + hcol];
        }
        float gi = acc[0 + s2][r] + xi;
        float gf = acc[2 + s2][r] + xf;
        float gc = acc[4 + s2][r] + xc;
        float go = acc[6 + s2][r] + xo;
        float cp = s2 ? c1[r] : c0[r];
        float cn = sigm(gf) * cp + sigm(gi) * tanhf(gc);
        float h  = sigm(go) * tanhf(cn);
        if (s2) c1[r] = cn; else c0[r] = cn;
        unsigned short hb = f2bf(h);
        // scatter h into A-layout LDS: (M,K=hcol) -> kt, lane, elem
        int kt2 = hcol >> 5, km = hcol & 31;
        int lHi = (km >> 3) & 1;
        int e   = (km & 7) + 8 * ((km >> 4) & 1);
        hpack[kt2 * 512 + ((M & 15) + 16 * lHi) * 16 + e] = hb;
        if (M < BB)
          hsOut[((size_t)(M * SS + tpos)) * outStride + colOff + hcol] = hb;
      }
    }
    __syncthreads();  // h fully updated before next step's reads
  }
}

// ---------------------------------------------------------------------------
// Final scores: one wave per batch row.
// ps = hsP[b, S-1, :2000(mapped)] . w2 + b2 ;  ws = hsW[b, S-1, :1000] . w1 + b1
// out[b] = wc0*ps + wc1*ws + bc
// ---------------------------------------------------------------------------
__global__ void k_final(const unsigned short* __restrict__ hsP,
                        const unsigned short* __restrict__ hsW,
                        const float* __restrict__ w1, const float* __restrict__ b1,
                        const float* __restrict__ w2, const float* __restrict__ b2,
                        const float* __restrict__ wc, const float* __restrict__ bc,
                        float* __restrict__ out) {
  const int lane = threadIdx.x & 31;
  const int b    = threadIdx.x >> 5;
  float ps = 0.f, ws = 0.f;
  const size_t rp = (size_t)(b * SS + SS - 1) * 2048;
  const size_t rw = (size_t)(b * SS + SS - 1) * 1024;
  for (int k = lane; k < 2000; k += 32) {
    int col = (k < 1000) ? k : 1024 + (k - 1000);
    ps += bf2f(hsP[rp + col]) * w2[k];
  }
  for (int k = lane; k < 1000; k += 32) ws += bf2f(hsW[rw + k]) * w1[k];
  for (int o = 16; o > 0; o >>= 1) {
    ps += __shfl_down(ps, o, 32);
    ws += __shfl_down(ws, o, 32);
  }
  if (lane == 0)
    out[b] = (ps + b2[0]) * wc[0] + (ws + b1[0]) * wc[1] + bc[0];
}

// ---------------------------------------------------------------------------
// Host orchestration.
// ---------------------------------------------------------------------------
extern "C" void kernel_launch(void* const* d_in, const int* in_sizes, int n_in,
                              void* d_out, int out_size, void* d_ws, size_t ws_size,
                              hipStream_t stream) {
  const int*   x        = (const int*)d_in[0];
  const float* emb_word = (const float*)d_in[1];
  const float* enc_emb  = (const float*)d_in[2];
  const float* l1_Wih0 = (const float*)d_in[3];
  const float* l1_Whh0 = (const float*)d_in[4];
  const float* l1_b0   = (const float*)d_in[5];
  const float* l1_Wih1 = (const float*)d_in[6];
  const float* l1_Whh1 = (const float*)d_in[7];
  const float* l1_b1   = (const float*)d_in[8];
  const float* f_Wih0  = (const float*)d_in[9];
  const float* f_Whh0  = (const float*)d_in[10];
  const float* f_b0    = (const float*)d_in[11];
  const float* r_Wih0  = (const float*)d_in[12];
  const float* r_Whh0  = (const float*)d_in[13];
  const float* r_b0    = (const float*)d_in[14];
  const float* f_Wih1  = (const float*)d_in[15];
  const float* f_Whh1  = (const float*)d_in[16];
  const float* f_b1    = (const float*)d_in[17];
  const float* r_Wih1  = (const float*)d_in[18];
  const float* r_Whh1  = (const float*)d_in[19];
  const float* r_b1    = (const float*)d_in[20];
  const float* w1 = (const float*)d_in[21];
  const float* b1 = (const float*)d_in[22];
  const float* w2 = (const float*)d_in[23];
  const float* b2 = (const float*)d_in[24];
  const float* wc = (const float*)d_in[25];
  const float* bc = (const float*)d_in[26];
  float* out = (float*)d_out;

  const int KT_E = 512 / 32, KT_P = 768 / 32, KT_H = 1024 / 32, KT_2H = 2048 / 32;

  size_t off = 0;
  auto alloc = [&](size_t bytes) -> void* {
    void* p = (char*)d_ws + off;
    off += (bytes + 255) & ~(size_t)255;
    return p;
  };
  auto btile = [&](int KT) { return (size_t)KT * NT * 512 * 2; };  // bf16 bytes

  // packed weights (bf16 B-tiles)
  unsigned short* Bih_l1_0 = (unsigned short*)alloc(btile(KT_E));
  unsigned short* Bhh_l1_0 = (unsigned short*)alloc(btile(KT_H));
  unsigned short* Bih_l1_1 = (unsigned short*)alloc(btile(KT_H));
  unsigned short* Bhh_l1_1 = (unsigned short*)alloc(btile(KT_H));
  unsigned short* Bih_f0   = (unsigned short*)alloc(btile(KT_P));
  unsigned short* Bih_r0   = (unsigned short*)alloc(btile(KT_P));
  unsigned short* Bhh_f0   = (unsigned short*)alloc(btile(KT_H));
  unsigned short* Bhh_r0   = (unsigned short*)alloc(btile(KT_H));
  unsigned short* Bih_f1   = (unsigned short*)alloc(btile(KT_2H));
  unsigned short* Bih_r1   = (unsigned short*)alloc(btile(KT_2H));
  unsigned short* Bhh_f1   = (unsigned short*)alloc(btile(KT_H));
  unsigned short* Bhh_r1   = (unsigned short*)alloc(btile(KT_H));
  // padded biases
  float* bias_l1_0 = (float*)alloc(NP * 4);
  float* bias_l1_1 = (float*)alloc(NP * 4);
  float* bias_f0   = (float*)alloc(NP * 4);
  float* bias_r0   = (float*)alloc(NP * 4);
  float* bias_f1   = (float*)alloc(NP * 4);
  float* bias_r1   = (float*)alloc(NP * 4);
  // activations (bf16 row-major)
  unsigned short* embW   = (unsigned short*)alloc((size_t)MM * 512 * 2);
  unsigned short* embP   = (unsigned short*)alloc((size_t)MM * 768 * 2);
  unsigned short* hsL1_0 = (unsigned short*)alloc((size_t)MM * 1024 * 2);
  unsigned short* hsL1_1 = (unsigned short*)alloc((size_t)MM * 1024 * 2);
  unsigned short* hsL2_0 = (unsigned short*)alloc((size_t)MM * 2048 * 2);
  unsigned short* hsL2_1 = (unsigned short*)alloc((size_t)MM * 2048 * 2);
  unsigned short* Apack  = (unsigned short*)alloc((size_t)MT * KT_2H * 512 * 2);
  float* xgF = (float*)alloc((size_t)MM * NP * 4);
  float* xgR = (float*)alloc((size_t)MM * NP * 4);
  (void)ws_size; (void)in_sizes; (void)n_in; (void)out_size;

  auto packB = [&](const float* W, unsigned short* dst, int Korig, int kO,
                   int kP, int KT) {
    int total = KT * NT * 512;
    k_pack_b<<<total / 256, 256, 0, stream>>>(W, dst, Korig, kO, kP, total);
  };
  auto packA = [&](const unsigned short* src, int Kp) {
    int KT = Kp / 32;
    int total = MT * KT * 512;
    k_pack_a<<<total / 256, 256, 0, stream>>>(src, Apack, Kp, KT, total);
  };
  auto gemm = [&](const unsigned short* Bp, const float* bias, float* xg, int KT) {
    k_gemm<<<dim3(NT / 8, MT / 4), 256, 0, stream>>>(Apack, Bp, bias, xg, KT);
  };

  // ---- pack all weights / biases (L2-resident afterwards) ----
  packB(l1_Wih0, Bih_l1_0, 512, 512, 512, KT_E);
  packB(l1_Whh0, Bhh_l1_0, 1000, 1000, 1024, KT_H);
  packB(l1_Wih1, Bih_l1_1, 1000, 1000, 1024, KT_H);
  packB(l1_Whh1, Bhh_l1_1, 1000, 1000, 1024, KT_H);
  packB(f_Wih0, Bih_f0, 768, 768, 768, KT_P);
  packB(r_Wih0, Bih_r0, 768, 768, 768, KT_P);
  packB(f_Whh0, Bhh_f0, 1000, 1000, 1024, KT_H);
  packB(r_Whh0, Bhh_r0, 1000, 1000, 1024, KT_H);
  packB(f_Wih1, Bih_f1, 2000, 1000, 1024, KT_2H);
  packB(r_Wih1, Bih_r1, 2000, 1000, 1024, KT_2H);
  packB(f_Whh1, Bhh_f1, 1000, 1000, 1024, KT_H);
  packB(r_Whh1, Bhh_r1, 1000, 1000, 1024, KT_H);
  k_pack_bias<<<NP / 256, 256, 0, stream>>>(l1_b0, bias_l1_0);
  k_pack_bias<<<NP / 256, 256, 0, stream>>>(l1_b1, bias_l1_1);
  k_pack_bias<<<NP / 256, 256, 0, stream>>>(f_b0, bias_f0);
  k_pack_bias<<<NP / 256, 256, 0, stream>>>(r_b0, bias_r0);
  k_pack_bias<<<NP / 256, 256, 0, stream>>>(f_b1, bias_f1);
  k_pack_bias<<<NP / 256, 256, 0, stream>>>(r_b1, bias_r1);

  // ---- word path (unidirectional x2) ----
  k_gather<<<dim3(2, MM), 256, 0, stream>>>(x, emb_word, embW, 512);
  packA(embW, 512);
  gemm(Bih_l1_0, bias_l1_0, xgF, KT_E);
  k_lstm<<<1, 1024, 0, stream>>>(xgF, xgF, Bhh_l1_0, Bhh_l1_0, hsL1_0, 1024);
  packA(hsL1_0, 1024);
  gemm(Bih_l1_1, bias_l1_1, xgF, KT_H);
  k_lstm<<<1, 1024, 0, stream>>>(xgF, xgF, Bhh_l1_1, Bhh_l1_1, hsL1_1, 1024);

  // ---- paragraph path (bidirectional x2) ----
  k_gather<<<dim3(3, MM), 256, 0, stream>>>(x, enc_emb, embP, 768);
  packA(embP, 768);
  gemm(Bih_f0, bias_f0, xgF, KT_P);
  gemm(Bih_r0, bias_r0, xgR, KT_P);
  k_lstm<<<2, 1024, 0, stream>>>(xgF, xgR, Bhh_f0, Bhh_r0, hsL2_0, 2048);
  packA(hsL2_0, 2048);
  gemm(Bih_f1, bias_f1, xgF, KT_2H);
  gemm(Bih_r1, bias_r1, xgR, KT_2H);
  k_lstm<<<2, 1024, 0, stream>>>(xgF, xgR, Bhh_f1, Bhh_r1, hsL2_1, 2048);

  // ---- final scores ----
  k_final<<<1, 256, 0, stream>>>(hsL2_1, hsL1_1, w1, b1, w2, b2, wc, bc, out);
}
